// PaiNN_66640712564908
// MI455X (gfx1250) — matrix operand order repaired and checked by reading, not compiled, not run on hardware
//
#include <hip/hip_runtime.h>
#include <hip/hip_bf16.h>
#include <math.h>

#define N_NODES 16384
#define N_EDGES 262144

typedef __attribute__((ext_vector_type(16))) __bf16 v16bf;
typedef __attribute__((ext_vector_type(8)))  float  v8f;

#define INV_SQRT_3f 0.57735026918962576f
#define INV_SQRT_Hf 0.0625f            /* 1/sqrt(256) */
#define INV_SQRT_2f 0.70710678118654752f

// silu(x)/0.6 with hardware reciprocal (v_rcp_f32) instead of IEEE division:
// denominator >= 1 so the approximation is well-conditioned.
__device__ __forceinline__ float ssilu_f(float x) {
  float r = __builtin_amdgcn_rcpf(1.0f + __expf(-x));
  return x * r * (1.0f / 0.6f);
}
__device__ __forceinline__ __bf16 f2bf(float f) { return (__bf16)f; }

// ---------------- WMMA fragment loaders (per CDNA5 ISA §7.12.2 layouts) ----
// A (16x32 bf16): lane<16 -> K = [hi*8 .. hi*8+7] and [16+hi*8 .. 16+hi*8+7]
// (two contiguous 16B chunks -> ds_load_b128 pair)

template <int S>
__device__ __forceinline__ v16bf load_afrag(const __bf16 (*sA)[S], int row, int k0, int hi) {
  v16bf a;
#pragma unroll
  for (int i = 0; i < 8; ++i) a[i] = sA[row][k0 + hi * 8 + i];
#pragma unroll
  for (int i = 8; i < 16; ++i) a[i] = sA[row][k0 + 8 + hi * 8 + i];
  return a;
}

// B (32x16 bf16) from TRANSPOSED LDS tile sBt[col][k]: lane's 16 K-values are
// two contiguous 8-chunks -> ds_load_b128 pair. kk = hi*8 + i (i<8), 8 + hi*8 + i (i>=8).
template <int S>
__device__ __forceinline__ v16bf load_bfrag_tr(const __bf16 (*sBt)[S], int col, int k0, int hi) {
  v16bf b;
#pragma unroll
  for (int i = 0; i < 16; ++i) b[i] = sBt[col][k0 + (i < 8 ? hi * 8 + i : 8 + hi * 8 + i)];
  return b;
}

// ---------------- Generic bf16-WMMA GEMM: C = act(A@B + bias) ------------
// A: M x K (row major, lda), B: K x N (row major, ldb), C: M x N (ldc)
// REQUIRES: M % 64 == 0, Nc % 128 == 0, K % 32 == 0 (true for all call sites).
// Block tile 64x128, BK=32; 256 threads = 8 waves in 2x4; wave tile 32x32.

__global__ __launch_bounds__(256) void gemm_bf16_wmma_kernel(
    const float* __restrict__ A, int lda,
    const float* __restrict__ B, int ldb,
    const float* __restrict__ bias,
    float* __restrict__ C, int ldc,
    int M, int Nc, int K, int act)
{
  __shared__ __bf16 sA[64][40];      // 64 rows x 32 K (stride 80B, 16B aligned)
  __shared__ __bf16 sBt[128][40];    // transposed: 128 cols x 32 K

  const int tid = threadIdx.x;
  const int lane = tid & 31;
  const int wave = tid >> 5;
  const int wm = wave >> 2;        // 0..1
  const int wn = wave & 3;         // 0..3
  const int hi = lane >> 4;
  const int l16 = lane & 15;

  const int row0 = blockIdx.y * 64;
  const int col0 = blockIdx.x * 128;

  const int bc = tid & 127;            // B: column owned by this thread
  const int brb = (tid >> 7) * 16;     // B: first K row

  v8f acc[2][2] = {};

  for (int k0 = 0; k0 < K; k0 += 32) {
    // A tile 64x32: 512 float4 loads (2 per thread), coalesced
#pragma unroll
    for (int i = 0; i < 2; ++i) {
      int q = tid + i * 256;
      int r = q >> 3, c4 = (q & 7) * 4;
      float4 v = *(const float4*)&A[(size_t)(row0 + r) * lda + k0 + c4];
      sA[r][c4 + 0] = f2bf(v.x);
      sA[r][c4 + 1] = f2bf(v.y);
      sA[r][c4 + 2] = f2bf(v.z);
      sA[r][c4 + 3] = f2bf(v.w);
    }
    // B tile 32x128 -> transposed store: per-K-row loads coalesced across
    // 128 threads; 16 contiguous bf16 stores per thread (b128-able)
    {
      const float* bp = &B[(size_t)(k0 + brb) * ldb + col0 + bc];
#pragma unroll
      for (int rr = 0; rr < 16; ++rr)
        sBt[bc][brb + rr] = f2bf(bp[(size_t)rr * ldb]);
    }
    __syncthreads();

#pragma unroll
    for (int mf = 0; mf < 2; ++mf) {
      v16bf af = load_afrag<40>(sA, wm * 32 + mf * 16 + l16, 0, hi);
#pragma unroll
      for (int nf = 0; nf < 2; ++nf) {
        v16bf bf = load_bfrag_tr<40>(sBt, wn * 32 + nf * 16 + l16, 0, hi);
        acc[mf][nf] = __builtin_amdgcn_wmma_f32_16x16x32_bf16(
            false, af, false, bf, (short)0, acc[mf][nf], false, false);
      }
    }
    __syncthreads();
  }

#pragma unroll
  for (int mf = 0; mf < 2; ++mf)
#pragma unroll
    for (int nf = 0; nf < 2; ++nf) {
      int gcol = col0 + wn * 32 + nf * 16 + l16;
      float bv = bias ? bias[gcol] : 0.0f;
#pragma unroll
      for (int r = 0; r < 8; ++r) {
        int grow = row0 + wm * 32 + mf * 16 + hi * 8 + r;
        float v = acc[mf][nf][r] + bv;
        if (act == 1) v = ssilu_f(v);
        C[(size_t)grow * ldc + gcol] = v;
      }
    }
}

// ---------------- Fused edge kernel ---------------------------------------
// Per block: 64 edges. rbfh = edge_rbf@W_rbf+b (WMMA, K=64) in 128-col chunks,
// m = rbfh * xh[src]; then scatter:
//   cols [0,256):    x_acc[dst] += m
//   cols [256,512):  v_acc[dst][d] += vec[src][d] * m * (1/sqrt3)*(1/sqrtH)
//   cols [512,768):  v_acc[dst][d] += m * edge_vector[d] * (1/sqrtH)
// REQUIRES: E % 64 == 0.

__global__ __launch_bounds__(256) void edge_message_kernel(
    const float* __restrict__ edge_rbf,    // E x 64
    const float* __restrict__ W_rbf,       // 64 x 768
    const float* __restrict__ b_rbf,       // 768
    const float* __restrict__ xh,          // N x 768
    const float* __restrict__ vec,         // N x 3 x 256
    const float* __restrict__ edge_vector, // E x 3
    const int*   __restrict__ edge_index,  // 2 x E
    float* __restrict__ x_acc,             // N x 256
    float* __restrict__ v_acc,             // N x 3 x 256
    int E_total)
{
  __shared__ __bf16 sA[64][72];     // rbf tile (64 edges x K=64), stride 144B
  __shared__ __bf16 sBt[128][72];   // transposed W_rbf tile (128 cols x K=64)
  __shared__ float  sC[64][132];    // result tile
  __shared__ float  sBias[768];
  __shared__ int    sSrc[64], sDst[64];
  __shared__ float  sEv[192];

  const int tid = threadIdx.x;
  const int lane = tid & 31;
  const int wave = tid >> 5;
  const int wm = wave >> 2;
  const int wn = wave & 3;
  const int hi = lane >> 4;
  const int l16 = lane & 15;
  const int e0 = blockIdx.x * 64;

#pragma unroll
  for (int i = 0; i < 4; ++i) {            // load rbf 64x64 as float4
    int q = tid + i * 256;
    int r = q >> 4, c4 = (q & 15) * 4;
    float4 v = *(const float4*)&edge_rbf[(size_t)(e0 + r) * 64 + c4];
    sA[r][c4 + 0] = f2bf(v.x);
    sA[r][c4 + 1] = f2bf(v.y);
    sA[r][c4 + 2] = f2bf(v.z);
    sA[r][c4 + 3] = f2bf(v.w);
  }
#pragma unroll
  for (int i = 0; i < 3; ++i) sBias[tid + i * 256] = b_rbf[tid + i * 256];
  if (tid < 64) {
    sSrc[tid] = edge_index[e0 + tid];
    sDst[tid] = edge_index[E_total + e0 + tid];
  }
  if (tid < 192) sEv[tid] = edge_vector[(size_t)e0 * 3 + tid];

  const int bc = tid & 127;
  const int brb = (tid >> 7) * 32;

  for (int chunk = 0; chunk < 6; ++chunk) {
    const int col0 = chunk * 128;
    __syncthreads();   // previous consumer done with sBt/sC
    // W_rbf tile 64x128 -> transposed store (32 K rows per thread)
    {
      const float* bp = &W_rbf[(size_t)brb * 768 + col0 + bc];
#pragma unroll
      for (int rr = 0; rr < 32; ++rr)
        sBt[bc][brb + rr] = f2bf(bp[(size_t)rr * 768]);
    }
    __syncthreads();

    v8f acc[2][2] = {};
#pragma unroll
    for (int ks = 0; ks < 64; ks += 32) {
#pragma unroll
      for (int mf = 0; mf < 2; ++mf) {
        v16bf af = load_afrag<72>(sA, wm * 32 + mf * 16 + l16, ks, hi);
#pragma unroll
        for (int nf = 0; nf < 2; ++nf) {
          v16bf bf = load_bfrag_tr<72>(sBt, wn * 32 + nf * 16 + l16, ks, hi);
          acc[mf][nf] = __builtin_amdgcn_wmma_f32_16x16x32_bf16(
              false, af, false, bf, (short)0, acc[mf][nf], false, false);
        }
      }
    }
#pragma unroll
    for (int mf = 0; mf < 2; ++mf)
#pragma unroll
      for (int nf = 0; nf < 2; ++nf)
#pragma unroll
        for (int r = 0; r < 8; ++r)
          sC[wm * 32 + mf * 16 + hi * 8 + r][wn * 32 + nf * 16 + l16] = acc[mf][nf][r];
    __syncthreads();

    // consume: 64 edges x 128 cols = 8192 elems, 32 per thread
#pragma unroll 4
    for (int it = 0; it < 32; ++it) {
      int idx = it * 256 + tid;
      int el = idx >> 7, cl = idx & 127;
      int gcol = col0 + cl;
      int src = sSrc[el], dst = sDst[el];
      float m = (sC[el][cl] + sBias[gcol]) * xh[(size_t)src * 768 + gcol];
      if (gcol < 256) {
        atomicAdd(&x_acc[(size_t)dst * 256 + gcol], m);
      } else if (gcol < 512) {
        int h = gcol - 256;
        float m2 = m * (INV_SQRT_3f * INV_SQRT_Hf);
#pragma unroll
        for (int d = 0; d < 3; ++d)
          atomicAdd(&v_acc[((size_t)dst * 3 + d) * 256 + h],
                    vec[((size_t)src * 3 + d) * 256 + h] * m2);
      } else {
        int h = gcol - 512;
        float m3 = m * INV_SQRT_Hf;
#pragma unroll
        for (int d = 0; d < 3; ++d)
          atomicAdd(&v_acc[((size_t)dst * 3 + d) * 256 + h], m3 * sEv[el * 3 + d]);
      }
    }
  }
}

// ---------------- Elementwise / reduction helper kernels -------------------

__global__ __launch_bounds__(256) void layernorm_kernel(
    const float* __restrict__ x, const float* __restrict__ g,
    const float* __restrict__ b, float* __restrict__ out, int rows)
{
  int gw = (blockIdx.x * 256 + threadIdx.x) >> 5;
  int lane = threadIdx.x & 31;
  if (gw >= rows) return;
  const float* row = x + (size_t)gw * 256;
  float vals[8];
  float s = 0.0f;
#pragma unroll
  for (int i = 0; i < 8; ++i) { vals[i] = row[lane + i * 32]; s += vals[i]; }
#pragma unroll
  for (int off = 16; off > 0; off >>= 1) s += __shfl_xor(s, off, 32);
  float mean = s * (1.0f / 256.0f);
  float vs = 0.0f;
#pragma unroll
  for (int i = 0; i < 8; ++i) { float d = vals[i] - mean; vs += d * d; }
#pragma unroll
  for (int off = 16; off > 0; off >>= 1) vs += __shfl_xor(vs, off, 32);
  float rstd = rsqrtf(vs * (1.0f / 256.0f) + 1e-5f);
  float* o = out + (size_t)gw * 256;
#pragma unroll
  for (int i = 0; i < 8; ++i) {
    int c = lane + i * 32;
    o[c] = (vals[i] - mean) * rstd * g[c] + b[c];
  }
}

__global__ void copy4_kernel(const float4* __restrict__ s, float4* __restrict__ d, size_t n4) {
  size_t i = blockIdx.x * (size_t)blockDim.x + threadIdx.x;
  if (i < n4) d[i] = s[i];
}

__global__ void vecdot_kernel(const float* __restrict__ vp, float* __restrict__ vec_dot, int nodes) {
  size_t idx = blockIdx.x * (size_t)blockDim.x + threadIdx.x;
  if (idx >= (size_t)nodes * 256) return;
  int n = idx >> 8, h = idx & 255;
  const float* base = vp + (size_t)n * 3 * 512;
  float s = base[h] * base[256 + h] + base[512 + h] * base[768 + h] +
            base[1024 + h] * base[1280 + h];
  vec_dot[idx] = s * INV_SQRT_Hf;
}

__global__ void norm3_kernel(const float* __restrict__ src, int src_ld, int src_off,
                             int cols, float eps, float* __restrict__ dst,
                             int dst_ld, int dst_off, int nodes)
{
  size_t idx = blockIdx.x * (size_t)blockDim.x + threadIdx.x;
  size_t total = (size_t)nodes * cols;
  if (idx >= total) return;
  int n = idx / cols, h = idx % cols;
  const float* b = src + (size_t)n * 3 * src_ld + src_off + h;
  float s = eps + b[0] * b[0] + b[src_ld] * b[src_ld] + b[2 * src_ld] * b[2 * src_ld];
  dst[(size_t)n * dst_ld + dst_off + h] = sqrtf(s);
}

__global__ void strided_copy_kernel(const float* __restrict__ src, int src_ld,
                                    float* __restrict__ dst, int dst_ld, int dst_off,
                                    int rows, int cols)
{
  size_t idx = blockIdx.x * (size_t)blockDim.x + threadIdx.x;
  size_t total = (size_t)rows * cols;
  if (idx >= total) return;
  int r = idx / cols, c = idx % cols;
  dst[(size_t)r * dst_ld + dst_off + c] = src[(size_t)r * src_ld + c];
}

__global__ void x_update_kernel(float* __restrict__ x, const float* __restrict__ xvh,
                                const float* __restrict__ vec_dot, int nodes)
{
  size_t idx = blockIdx.x * (size_t)blockDim.x + threadIdx.x;
  if (idx >= (size_t)nodes * 256) return;
  int n = idx >> 8, h = idx & 255;
  float xv1 = xvh[(size_t)n * 768 + h];
  float xv2 = xvh[(size_t)n * 768 + 256 + h];
  x[idx] += (xv1 + xv2 * vec_dot[idx]) * INV_SQRT_2f;
}

__global__ void v_update_kernel(float* __restrict__ v, const float* __restrict__ xvh,
                                const float* __restrict__ vp, int nodes)
{
  size_t idx = blockIdx.x * (size_t)blockDim.x + threadIdx.x;
  if (idx >= (size_t)nodes * 768) return;
  int n = idx / 768, rem = idx % 768;
  int d = rem >> 8, h = rem & 255;
  v[idx] += xvh[(size_t)n * 768 + 512 + h] * vp[((size_t)n * 3 + d) * 512 + h];
}

__global__ void gated_epi_kernel(const float* __restrict__ hh, const float* __restrict__ vec2,
                                 float* __restrict__ x_out, float* __restrict__ v_out,
                                 int C, int nodes)
{
  size_t idx = blockIdx.x * (size_t)blockDim.x + threadIdx.x;
  size_t nx = (size_t)nodes * C;
  size_t nv = nx * 3;
  if (idx < nx) {
    int n = idx / C, h = idx % C;
    x_out[idx] = ssilu_f(hh[(size_t)n * 2 * C + h]);
  }
  if (idx < nv) {
    int n = idx / (3 * C);
    int h = (idx % (3 * C)) % C;
    v_out[idx] = hh[(size_t)n * 2 * C + C + h] * vec2[idx];
  }
}

__global__ __launch_bounds__(256) void gemv_kernel(
    const float* __restrict__ A, int lda, const float* __restrict__ B, int ldb,
    const float* __restrict__ bias, float* __restrict__ out, int ldo,
    int rows, int K, int ncols)
{
  int gw = (blockIdx.x * 256 + threadIdx.x) >> 5;
  int lane = threadIdx.x & 31;
  if (gw >= rows) return;
  for (int c = 0; c < ncols; ++c) {
    float s = 0.0f;
    for (int k = lane; k < K; k += 32) s += A[(size_t)gw * lda + k] * B[(size_t)k * ldb + c];
#pragma unroll
    for (int off = 16; off > 0; off >>= 1) s += __shfl_xor(s, off, 32);
    if (lane == 0) out[(size_t)gw * ldo + c] = s + (bias ? bias[c] : 0.0f);
  }
}

__global__ void final_out_kernel(const float* __restrict__ hh2, const float* __restrict__ vec2g2,
                                 float* __restrict__ out, int nodes)
{
  size_t idx = blockIdx.x * (size_t)blockDim.x + threadIdx.x;
  if (idx >= (size_t)nodes * 3) return;
  int n = idx / 3;
  out[idx] = hh2[(size_t)n * 2 + 1] * vec2g2[idx];
}

// ---------------- Host orchestration --------------------------------------

extern "C" void kernel_launch(void* const* d_in, const int* in_sizes, int n_in,
                              void* d_out, int out_size, void* d_ws, size_t ws_size,
                              hipStream_t stream)
{
  (void)in_sizes; (void)n_in; (void)out_size; (void)ws_size;
  const float* x           = (const float*)d_in[0];
  const float* vec         = (const float*)d_in[1];
  const float* edge_rbf    = (const float*)d_in[2];
  const float* edge_vector = (const float*)d_in[3];
  const float* ln_g        = (const float*)d_in[4];
  const float* ln_b        = (const float*)d_in[5];
  const float* W_x1        = (const float*)d_in[6];
  const float* b_x1        = (const float*)d_in[7];
  const float* W_x2        = (const float*)d_in[8];
  const float* b_x2        = (const float*)d_in[9];
  const float* W_rbf       = (const float*)d_in[10];
  const float* b_rbf       = (const float*)d_in[11];
  const float* W_vp        = (const float*)d_in[12];
  const float* W_xv1       = (const float*)d_in[13];
  const float* b_xv1       = (const float*)d_in[14];
  const float* W_xv2       = (const float*)d_in[15];
  const float* b_xv2       = (const float*)d_in[16];
  const float* o1_Wv1      = (const float*)d_in[17];
  const float* o1_Wv2      = (const float*)d_in[18];
  const float* o1_Wu1      = (const float*)d_in[19];
  const float* o1_bu1      = (const float*)d_in[20];
  const float* o1_Wu2      = (const float*)d_in[21];
  const float* o1_bu2      = (const float*)d_in[22];
  const float* o2_Wv1      = (const float*)d_in[23];
  const float* o2_Wv2      = (const float*)d_in[24];
  const float* o2_Wu1      = (const float*)d_in[25];
  const float* o2_bu1      = (const float*)d_in[26];
  const float* o2_Wu2      = (const float*)d_in[27];
  const float* o2_bu2      = (const float*)d_in[28];
  const int*   edge_index  = (const int*)d_in[29];
  float* out = (float*)d_out;

  const size_t N = N_NODES;
  const int    E = N_EDGES;
  float* ws = (float*)d_ws;
  // workspace layout (floats), aggressively reused
  float* ln_x    = ws;                    // N*256
  float* h1      = ln_x + N * 256;        // N*256  (also h2, hg, hg2)
  float* xh      = h1 + N * 256;          // N*768  (also xvh, then x_c/v_c)
  float* x_a     = xh + N * 768;          // N*256
  float* v_a     = x_a + N * 256;         // N*768
  float* vp      = v_a + N * 768;         // 3N*512 (also t1/t2 at base, vec2g at +3N*256)
  float* vec_dot = vp + N * 3 * 512;      // N*256
  float* cat     = vec_dot + N * 256;     // N*512
  float* vec2g2  = cat + N * 512;         // 3N
  float* hh2     = vec2g2 + N * 3;        // 2N
  float* t1    = vp;
  float* t2    = vp;
  float* vec2g = vp + N * 3 * 256;
  float* hh    = ln_x;
  float* x_c   = xh;
  float* v_c   = xh + N * 128;

  auto GEMM = [&](const float* A, int lda, const float* B, int ldb, const float* bias,
                  float* C, int ldc, int M, int Nc, int K, int act) {
    dim3 g((Nc + 127) / 128, (M + 63) / 64);
    gemm_bf16_wmma_kernel<<<g, dim3(256), 0, stream>>>(A, lda, B, ldb, bias, C, ldc, M, Nc, K, act);
  };
  auto EW = [&](size_t total) { return dim3((unsigned)((total + 255) / 256)); };

  // ---- node MLP: xh = ssilu(LN(x)@W_x1+b)@W_x2+b ----
  layernorm_kernel<<<dim3(N / 8), dim3(256), 0, stream>>>(x, ln_g, ln_b, ln_x, (int)N);
  GEMM(ln_x, 256, W_x1, 256, b_x1, h1, 256, (int)N, 256, 256, 1);
  GEMM(h1, 256, W_x2, 768, b_x2, xh, 768, (int)N, 768, 256, 0);

  // ---- init accumulators, fused edge message + scatter ----
  copy4_kernel<<<EW(N * 64), dim3(256), 0, stream>>>((const float4*)x, (float4*)x_a, N * 64);
  copy4_kernel<<<EW(N * 192), dim3(256), 0, stream>>>((const float4*)vec, (float4*)v_a, N * 192);
  edge_message_kernel<<<dim3(E / 64), dim3(256), 0, stream>>>(
      edge_rbf, W_rbf, b_rbf, xh, vec, edge_vector, edge_index, x_a, v_a, E);

  // ---- vector projection + mixing ----
  GEMM(v_a, 256, W_vp, 512, nullptr, vp, 512, (int)(3 * N), 512, 256, 0);
  vecdot_kernel<<<EW(N * 256), dim3(256), 0, stream>>>(vp, vec_dot, (int)N);
  norm3_kernel<<<EW(N * 256), dim3(256), 0, stream>>>(vp, 512, 256, 256, 1e-8f, cat, 512, 256, (int)N);
  strided_copy_kernel<<<EW(N * 256), dim3(256), 0, stream>>>(x_a, 256, cat, 512, 0, (int)N, 256);
  GEMM(cat, 512, W_xv1, 256, b_xv1, h1, 256, (int)N, 256, 512, 1);
  GEMM(h1, 256, W_xv2, 768, b_xv2, xh, 768, (int)N, 768, 256, 0);
  x_update_kernel<<<EW(N * 256), dim3(256), 0, stream>>>(x_a, xh, vec_dot, (int)N);
  v_update_kernel<<<EW(N * 768), dim3(256), 0, stream>>>(v_a, xh, vp, (int)N);

  // ---- gated equivariant block 1 (out_ch = 128) ----
  GEMM(v_a, 256, o1_Wv1, 256, nullptr, t1, 256, (int)(3 * N), 256, 256, 0);
  GEMM(v_a, 256, o1_Wv2, 128, nullptr, vec2g, 128, (int)(3 * N), 128, 256, 0);
  norm3_kernel<<<EW(N * 256), dim3(256), 0, stream>>>(t1, 256, 0, 256, 0.0f, cat, 512, 256, (int)N);
  strided_copy_kernel<<<EW(N * 256), dim3(256), 0, stream>>>(x_a, 256, cat, 512, 0, (int)N, 256);
  GEMM(cat, 512, o1_Wu1, 256, o1_bu1, h1, 256, (int)N, 256, 512, 1);
  GEMM(h1, 256, o1_Wu2, 256, o1_bu2, hh, 256, (int)N, 256, 256, 0);
  gated_epi_kernel<<<EW(N * 3 * 128), dim3(256), 0, stream>>>(hh, vec2g, x_c, v_c, 128, (int)N);

  // ---- gated equivariant block 2 (out_ch = 1) ----
  GEMM(v_c, 128, o2_Wv1, 128, nullptr, t2, 128, (int)(3 * N), 128, 128, 0);
  gemv_kernel<<<EW(3 * N * 32), dim3(256), 0, stream>>>(v_c, 128, o2_Wv2, 1, nullptr,
                                                        vec2g2, 1, (int)(3 * N), 128, 1);
  norm3_kernel<<<EW(N * 128), dim3(256), 0, stream>>>(t2, 128, 0, 128, 0.0f, cat, 256, 128, (int)N);
  strided_copy_kernel<<<EW(N * 128), dim3(256), 0, stream>>>(x_c, 128, cat, 256, 0, (int)N, 128);
  GEMM(cat, 256, o2_Wu1, 128, o2_bu1, h1, 128, (int)N, 128, 256, 1);
  gemv_kernel<<<EW(N * 32), dim3(256), 0, stream>>>(h1, 128, o2_Wu2, 2, o2_bu2,
                                                    hh2, 2, (int)N, 128, 2);
  final_out_kernel<<<EW(N * 3), dim3(256), 0, stream>>>(hh2, vec2g2, out, (int)N);
}
